// RelativeCoordConv_48009144434960
// MI455X (gfx1250) — matrix operand order repaired and checked by reading, compiled, not run on hardware
//
#include <hip/hip_runtime.h>
#include <hip/hip_bf16.h>

// ---------------- problem constants (match reference) ----------------
#define N_POINTS  400000
#define IN_CH     64
#define OUT_CH    64
#define K_VOL     27
#define CH_REAL   67
// feature row: 64 bf16 features (128B) + 4 f32 (x,y,z,0 coords) (16B)
#define ROW_B     144
// LDS layout: bf16 B fragments then f32 coord-B fragments
#define LDS_B16   (K_VOL * 2 * 4 * 1024)        // 221184 B
#define LDS_F32   (K_VOL * 4 * 256)             // 27648 B
#define LDS_TOT   (LDS_B16 + LDS_F32)           // 248832 B (fits 320KB/WGP)

typedef __attribute__((ext_vector_type(16))) __bf16 v16bf;
typedef __attribute__((ext_vector_type(8)))  float  v8f;
typedef __attribute__((ext_vector_type(4)))  int    v4i;
typedef __attribute__((ext_vector_type(2)))  float  v2f;

union FragU { v16bf bf; v4i q[2]; };

static __device__ __forceinline__ unsigned short f2bf(float f) {
    unsigned u = __builtin_bit_cast(unsigned, f);
    u += 0x7FFFu + ((u >> 16) & 1u);
    return (unsigned short)(u >> 16);
}

// ------------------------------------------------------------------
// Prepass 1: packed feature rows, (N_POINTS+1) x 144B.
//   [0,128): 64 bf16 features; [128,144): f32 {nx,ny,nz,0}.
//   row N_POINTS = zeros (reference pad row).
// grid = N_POINTS+1 blocks x 64 threads.
// ------------------------------------------------------------------
__global__ void prep_features(const float* __restrict__ features,
                              const float* __restrict__ point,
                              const int*   __restrict__ coords,
                              char* __restrict__ feat) {
    const int row = blockIdx.x;
    const int t   = threadIdx.x;              // 0..63
    char* rp = feat + (size_t)row * ROW_B;
    float v = (row < N_POINTS) ? features[(size_t)row * IN_CH + t] : 0.0f;
    ((unsigned short*)rp)[t] = f2bf(v);
    if (t < 4) {
        float c = 0.0f;
        if (row < N_POINTS && t < 3)
            c = ((float)coords[row * 3 + t] - point[t]) * (1.0f / 256.0f);
        ((float*)(rp + 128))[t] = c;
    }
}

// ------------------------------------------------------------------
// Prepass 2a: bf16 B fragments (feature channels 0..63) in the CDNA5
// 16-bit B (32x16) register layout.  f = ((k*2 + c)*4 + nt), 216 frags.
// ------------------------------------------------------------------
__global__ void prep_wfrag(const float* __restrict__ W,
                           unsigned int* __restrict__ Bglob) {
    const int f    = blockIdx.x;              // 0..215
    const int lane = threadIdx.x;             // 0..31
    const int nt   = f & 3;
    const int c    = (f >> 2) & 1;
    const int k    = f >> 3;
    const int n    = nt * 16 + (lane & 15);
    const int half = lane >> 4;
    unsigned int* dst = Bglob + (size_t)f * 256 + lane * 8;
#pragma unroll
    for (int j = 0; j < 8; ++j) {
        const int k0 = c * 32 + (j >> 2) * 16 + half * 8 + 2 * (j & 3);      // <= 63 < CH_REAL
        const float v0 = W[((size_t)k * CH_REAL + k0    ) * OUT_CH + n];
        const float v1 = W[((size_t)k * CH_REAL + k0 + 1) * OUT_CH + n];
        dst[j] = (unsigned)f2bf(v0) | ((unsigned)f2bf(v1) << 16);
    }
}

// ------------------------------------------------------------------
// Prepass 2b: f32 coord-B fragments (W rows 64..66, padded K=4) in the
// 32-bit B (4x16) layout: lanes 0-15 -> K=j, lanes 16-31 -> K=2+j.
// g = k*4 + nt, 108 fragments of 256B, placed after the bf16 region.
// ------------------------------------------------------------------
__global__ void prep_wcoord(const float* __restrict__ W,
                            char* __restrict__ Bglob) {
    const int g    = blockIdx.x;              // 0..107
    const int lane = threadIdx.x;             // 0..31
    const int k    = g >> 2;
    const int nt   = g & 3;
    const int n    = nt * 16 + (lane & 15);
    const int half = lane >> 4;
    float* dst = (float*)(Bglob + LDS_B16) + (size_t)g * 64 + lane * 2;
#pragma unroll
    for (int j = 0; j < 2; ++j) {
        const int kr = half * 2 + j;          // 0..3
        dst[j] = (kr < 3) ? W[((size_t)k * CH_REAL + 64 + kr) * OUT_CH + n] : 0.0f;
    }
}

// ------------------------------------------------------------------
// Main gather-GEMM.  512 threads = 16 waves; each wave: two 16-row
// M-tiles x four 16-col N-tiles.  All B fragments (243KB) are staged
// in LDS once via async global->LDS DMA, then per k:
//   16 x v_wmma_f32_16x16x32_bf16 (feature channels)
//  + 8 x v_wmma_f32_16x16x4_f32   (exact-fp32 coord channels)
// ------------------------------------------------------------------
__global__ void __launch_bounds__(512)
sparse_conv_wmma(const char* __restrict__ feat,
                 const char* __restrict__ Bglob,
                 const int*  __restrict__ nbr,
                 float*      __restrict__ out) {
    extern __shared__ char smem[];

    const int tid   = threadIdx.x;
    const int lane  = tid & 31;
    const int wave  = tid >> 5;               // 0..15
    const int lm    = lane & 15;
    const int half  = lane >> 4;
    const int mbase = blockIdx.x * 512 + wave * 32;

    // ---- one-time cooperative preload of all B fragments into LDS ----
    {
        const unsigned sbase = (unsigned)(size_t)(void*)smem;  // LDS offset
        for (int i = tid * 16; i < LDS_TOT; i += 512 * 16) {
            const unsigned long long ga = (unsigned long long)(Bglob + i);
            const unsigned la = sbase + (unsigned)i;
            asm volatile("global_load_async_to_lds_b128 %0, %1, off"
                         :: "v"(la), "v"(ga) : "memory");
        }
        asm volatile("s_wait_asynccnt 0x0" ::: "memory");
    }
    __syncthreads();

    v8f acc[2][4];
#pragma unroll
    for (int t = 0; t < 2; ++t)
#pragma unroll
        for (int nt = 0; nt < 4; ++nt) acc[t][nt] = {};

    const int m0r = mbase + lm;
    const int m1r = mbase + 16 + lm;

    for (int k = 0; k < K_VOL; ++k) {
        const int* nrow = nbr + (size_t)k * N_POINTS;
        const int r0 = (m0r < N_POINTS) ? nrow[m0r] : N_POINTS;
        const int r1 = (m1r < N_POINTS) ? nrow[m1r] : N_POINTS;
        const char* p0 = feat + (size_t)r0 * ROW_B;
        const char* p1 = feat + (size_t)r1 * ROW_B;

        // fp32 coord A fragments (16x4): lane<16 -> {x,y}, lane>=16 -> {z,0}
        const v2f a0c = *(const v2f*)(p0 + 128 + half * 8);
        const v2f a1c = *(const v2f*)(p1 + 128 + half * 8);

#pragma unroll
        for (int c = 0; c < 2; ++c) {
            // bf16 A (16x32): lane<16 -> K c*32+{0..7}@+0, {16..23}@+32;
            // lane>=16 -> complementary 16B blocks.
            const int off = c * 64 + half * 16;
            FragU a0, a1;
            a0.q[0] = *(const v4i*)(p0 + off);
            a0.q[1] = *(const v4i*)(p0 + off + 32);
            a1.q[0] = *(const v4i*)(p1 + off);
            a1.q[1] = *(const v4i*)(p1 + off + 32);
            const char* bb = smem + (size_t)((k * 2 + c) * 4) * 1024 + lane * 32;
#pragma unroll
            for (int nt = 0; nt < 4; ++nt) {
                FragU b;
                b.q[0] = *(const v4i*)(bb + nt * 1024);
                b.q[1] = *(const v4i*)(bb + nt * 1024 + 16);
                acc[0][nt] = __builtin_amdgcn_wmma_f32_16x16x32_bf16(
                    false, a0.bf, false, b.bf, (short)0, acc[0][nt], false, false);
                acc[1][nt] = __builtin_amdgcn_wmma_f32_16x16x32_bf16(
                    false, a1.bf, false, b.bf, (short)0, acc[1][nt], false, false);
            }
        }

        // fp32 coord contribution: K=4 WMMA, B from LDS
        const char* cb = smem + LDS_B16 + (size_t)(k * 4) * 256 + lane * 8;
#pragma unroll
        for (int nt = 0; nt < 4; ++nt) {
            const v2f bc = *(const v2f*)(cb + nt * 256);
            acc[0][nt] = __builtin_amdgcn_wmma_f32_16x16x4_f32(
                false, a0c, false, bc, (short)0, acc[0][nt], false, false);
            acc[1][nt] = __builtin_amdgcn_wmma_f32_16x16x4_f32(
                false, a1c, false, bc, (short)0, acc[1][nt], false, false);
        }
    }

    // C layout: VGPR v -> M = v + 8*half, N = lm + 16*nt.
#pragma unroll
    for (int t = 0; t < 2; ++t)
#pragma unroll
        for (int nt = 0; nt < 4; ++nt)
#pragma unroll
            for (int v = 0; v < 8; ++v) {
                const int m = mbase + t * 16 + half * 8 + v;
                if (m < N_POINTS)
                    out[(size_t)m * OUT_CH + nt * 16 + lm] = acc[t][nt][v];
            }
}

// ------------------------------------------------------------------
extern "C" void kernel_launch(void* const* d_in, const int* in_sizes, int n_in,
                              void* d_out, int out_size, void* d_ws, size_t ws_size,
                              hipStream_t stream) {
    const float* features = (const float*)d_in[0];   // 400000 x 64 f32
    const float* point    = (const float*)d_in[1];   // 3 f32
    const float* W        = (const float*)d_in[2];   // 27 x 67 x 64 f32
    const int*   coords   = (const int*)d_in[3];     // 400000 x 3 i32
    const int*   nbr      = (const int*)d_in[4];     // 27 x 400000 i32
    float* out = (float*)d_out;

    unsigned char* ws = (unsigned char*)d_ws;
    char* feat = (char*)ws;                                  // (N+1)*144B ~ 57.6MB
    size_t feat_bytes = ((size_t)(N_POINTS + 1) * ROW_B + 255) & ~(size_t)255;
    char* Bglob = (char*)(ws + feat_bytes);                  // 248832 B fragments

    prep_features<<<N_POINTS + 1, 64, 0, stream>>>(features, point, coords, feat);
    prep_wfrag<<<K_VOL * 2 * 4, 32, 0, stream>>>(W, (unsigned int*)Bglob);
    prep_wcoord<<<K_VOL * 4, 32, 0, stream>>>(W, Bglob);

    const int grid = (N_POINTS + 511) / 512;                 // 782
    sparse_conv_wmma<<<grid, 512, LDS_TOT, stream>>>(feat, Bglob, nbr, out);
}
